// Architecture_7301444403346
// MI455X (gfx1250) — compile-verified, matrix operand order checked by reading, not run on hardware
//
#include <hip/hip_runtime.h>
#include <hip/hip_bf16.h>
#include <math.h>

// ---------------------------------------------------------------------------
// Model constants (from the reference)
// ---------------------------------------------------------------------------
#define BB   8
#define DD   256
#define LL   1024
#define KK   24
#define KU   3
#define KYY  2
#define NL   6
#define DT   10
#define BL   (BB * LL)        // 8192 rows
#define KD   (KK * DD)        // 6144
#define NFFT 2048
#define EPSV 1e-5f

typedef __attribute__((ext_vector_type(16))) __bf16    v16bf;
typedef __attribute__((ext_vector_type(8)))  __bf16    bf16x8;
typedef __attribute__((ext_vector_type(8)))  float     v8f;
typedef __attribute__((ext_vector_type(4)))  unsigned  u32x4;
typedef __attribute__((ext_vector_type(8)))  unsigned  u32x8;

__device__ inline __bf16 f2bf(float f) {
    unsigned u = __builtin_bit_cast(unsigned, f);
    unsigned r = (u + 0x7FFFu + ((u >> 16) & 1u)) >> 16;
    return __builtin_bit_cast(__bf16, (unsigned short)r);
}

__device__ inline unsigned pack2bf(float lo, float hi) {
    return (unsigned)__builtin_bit_cast(unsigned short, f2bf(lo)) |
           ((unsigned)__builtin_bit_cast(unsigned short, f2bf(hi)) << 16);
}

// Swizzled position of k (0..31) inside a B-fragment-ordered 32-element row
__device__ inline int swzk(int k) {
    return ((k >> 3) & 1) * 16 + ((k >> 4) & 1) * 8 + (k & 7);
}

__device__ inline v16bf cat16(bf16x8 lo, bf16x8 hi) {
    return __builtin_shufflevector(lo, hi, 0, 1, 2, 3, 4, 5, 6, 7,
                                   8, 9, 10, 11, 12, 13, 14, 15);
}

// A-fragment from a row-major (by k) 32-elem LDS row: ks {khalf..+7, 16+khalf..+7}
__device__ inline v16bf load_afrag(const __bf16* row, int khalf) {
    bf16x8 lo = *(const bf16x8*)(row + khalf);
    bf16x8 hi = *(const bf16x8*)(row + 16 + khalf);
    return cat16(lo, hi);
}

// B-fragment from a swizzled 32-elem LDS row: 16 contiguous elems at half*16
__device__ inline v16bf load_bfrag(const __bf16* row, int half) {
    bf16x8 lo = *(const bf16x8*)(row + half * 16);
    bf16x8 hi = *(const bf16x8*)(row + half * 16 + 8);
    return cat16(lo, hi);
}

// ---------------------------------------------------------------------------
// Misc small kernels
// ---------------------------------------------------------------------------
__global__ __launch_bounds__(256) void zero_f32(float* p, int n) {
    int i = blockIdx.x * 256 + threadIdx.x;
    if (i < n) p[i] = 0.0f;
}

__global__ __launch_bounds__(256) void f32_to_bf16(__bf16* dst, const float* src, int n) {
    int i = blockIdx.x * 256 + threadIdx.x;
    if (i < n) dst[i] = f2bf(src[i]);
}

// inputs [B,3,1024] -> x [BL, D]
__global__ __launch_bounds__(256)
void embed_kernel(const float* in, const float* w, const float* bias, float* x) {
    int row = blockIdx.x;
    int d   = threadIdx.x;
    int b = row >> 10, l = row & 1023;
    float acc = bias[d];
    #pragma unroll
    for (int c = 0; c < 3; ++c)
        acc += in[b * 3 * LL + c * LL + l] * w[c * DD + d];
    x[(size_t)row * DD + d] = acc;
}

// ---------------------------------------------------------------------------
// BatchNorm
// ---------------------------------------------------------------------------
__global__ __launch_bounds__(256)
void bn_stats(const float* x, float* stats) {
    int t = threadIdx.x;
    int r0 = blockIdx.x * 256;
    float s = 0.f, ss = 0.f;
    for (int r = 0; r < 256; ++r) {
        float v = x[(size_t)(r0 + r) * DD + t];
        s += v; ss += v * v;
    }
    atomicAdd(&stats[t], s);
    atomicAdd(&stats[DD + t], ss);
}

__global__ __launch_bounds__(256)
void bn_apply(const float* x, const float* stats, const float* gamma, const float* beta,
              float* xn) {
    int row = blockIdx.x, d = threadIdx.x;
    float mean = stats[d] * (1.0f / (float)BL);
    float var  = stats[DD + d] * (1.0f / (float)BL) - mean * mean;
    float v = (x[(size_t)row * DD + d] - mean) * rsqrtf(var + EPSV) * gamma[d] + beta[d];
    xn[(size_t)row * DD + d] = v;
}

// ---------------------------------------------------------------------------
// In-LDS radix-2 Stockham FFT, length 2048, 256 threads.
// ---------------------------------------------------------------------------
__device__ float2* fft2048(float2* s0, float2* s1, int tid, float sign) {
    float2 *src = s0, *dst = s1;
    int l = NFFT >> 1, m = 1;
    while (l >= 1) {
        __syncthreads();
        for (int i = tid; i < (NFFT >> 1); i += 256) {
            int j = i / m;
            int k = i - j * m;
            float ang = sign * 3.14159265358979f * (float)j / (float)l;
            float sn, cs;
            __sincosf(ang, &sn, &cs);
            float2 c0 = src[k + j * m];
            float2 c1 = src[k + j * m + (NFFT >> 1)];
            float2 d0 = {c0.x + c1.x, c0.y + c1.y};
            float2 e  = {c0.x - c1.x, c0.y - c1.y};
            float2 d1 = {cs * e.x - sn * e.y, sn * e.x + cs * e.y};
            dst[k + 2 * j * m]     = d0;
            dst[k + 2 * j * m + m] = d1;
        }
        float2* t = src; src = dst; dst = t;
        l >>= 1; m <<= 1;
    }
    __syncthreads();
    return src;
}

__global__ __launch_bounds__(256)
void fft_filters(const float* eig_vecs, const float* eig_vals, float2* Vf) {
    __shared__ float2 s0[NFFT], s1[NFFT];
    int k = blockIdx.x, tid = threadIdx.x;
    for (int i = tid; i < NFFT; i += 256) {
        float v = (i < LL) ? eig_vecs[i * KK + k] : 0.f;
        s0[i] = make_float2(v, 0.f);
    }
    float2* res = fft2048(s0, s1, tid, -1.f);
    float sc = __powf(eig_vals[k], 0.25f);
    for (int i = tid; i < NFFT; i += 256)
        Vf[(size_t)k * NFFT + i] = make_float2(res[i].x * sc, res[i].y * sc);
}

__global__ __launch_bounds__(256)
void fft_forward(const float* xn, float2* U) {
    __shared__ float2 s0[NFFT], s1[NFFT];
    int bd = blockIdx.x, tid = threadIdx.x;
    int b = bd >> 8, d = bd & 255;
    for (int i = tid; i < NFFT; i += 256) {
        float v = (i < LL) ? xn[(size_t)(b * LL + i) * DD + d] : 0.f;
        s0[i] = make_float2(v, 0.f);
    }
    float2* res = fft2048(s0, s1, tid, -1.f);
    for (int i = tid; i < NFFT; i += 256)
        U[(size_t)bd * NFFT + i] = res[i];
}

// freq product + inverse FFT -> x~ stored TRANSPOSED: xtT[(k*D+d)*BL + row]
__global__ __launch_bounds__(256)
void conv_ifft(const float2* Vf, const float2* U, __bf16* xtT) {
    __shared__ float2 s0[NFFT], s1[NFFT];
    int id = blockIdx.x, tid = threadIdx.x;
    int d = id & 255;
    int k = (id >> 8) % KK;
    int b = id / (256 * KK);
    const float2* v = Vf + (size_t)k * NFFT;
    const float2* u = U + (size_t)(b * DD + d) * NFFT;
    for (int i = tid; i < NFFT; i += 256) {
        float2 a = v[i], c = u[i];
        s0[i] = make_float2(a.x * c.x - a.y * c.y, a.x * c.y + a.y * c.x);
    }
    float2* res = fft2048(s0, s1, tid, +1.f);
    const float inv = 1.0f / (float)NFFT;
    __bf16* dst = xtT + (size_t)(k * DD + d) * BL + b * LL;
    for (int i = tid; i < LL; i += 256)
        dst[i] = f2bf(res[i].x * inv);
}

// ---------------------------------------------------------------------------
// WMMA bf16 GEMM, double-buffered LDS, optional split-K.
//  EPI: 0=store, 1=load+add+store, 2=bias+store, 3=atomicAdd (split-K)
//  ATRANS=false: A row-major [M,Kd]; A tile DMA'd by the Tensor Data Mover
//    (2D tile 32x128, 2B elems, pad 16B per 64B row -> 80B LDS row stride),
//    pipelined one tile ahead and synced with s_wait_tensorcnt.
//  ATRANS=true : A is [Kd,M]; coalesced read + LDS transpose.
// ---------------------------------------------------------------------------
template <bool ATRANS, int EPI>
__global__ __launch_bounds__(256)
void gemm_bf16(const __bf16* __restrict__ A, const __bf16* __restrict__ Bm,
               float* __restrict__ C, const float* __restrict__ bias,
               int M, int N, int Kd, int kChunk) {
    __shared__ __bf16 As[2][128][40];   // 80B row stride: 16B aligned, conflict-free frags
    __shared__ __bf16 Bsw[2][128][40];  // swizzled: Bsw[.][n][swzk(k)]
    int tid  = threadIdx.x;
    int lane = tid & 31;
    int wave = tid >> 5;
    int waveM = (wave >> 2) * 64;
    int waveN = (wave & 3) * 32;
    int mBase = blockIdx.y * 128;
    int nBase = blockIdx.x * 128;
    int mlane = lane & 15;
    int half  = lane >> 4;
    int khalf = half * 8;
    int kStart = blockIdx.z * kChunk;
    int nT = kChunk / 32;

    auto stage = [&](int k0, int buf) {
        if (!ATRANS) {
            if (wave == 0) {   // Tensor Data Mover: A tile [32 k x 128 m]
                unsigned ldsA = (unsigned)(unsigned long long)&As[buf][0][0];
                unsigned long long ga =
                    (unsigned long long)(A + (size_t)mBase * Kd + k0);
                u32x4 g0;
                g0[0] = 1u;
                g0[1] = ldsA;
                g0[2] = (unsigned)ga;
                g0[3] = (unsigned)((ga >> 32) & 0x1FFFFFFu) | (2u << 30);
                u32x8 g1;
                g1[0] = (1u << 16) | (1u << 20) | (3u << 22) | (3u << 25);
                unsigned td0 = (unsigned)Kd, td1 = (unsigned)M;
                g1[1] = (td0 & 0xffffu) << 16;
                g1[2] = ((td0 >> 16) & 0xffffu) | ((td1 & 0xffffu) << 16);
                g1[3] = ((td1 >> 16) & 0xffffu) | (32u << 16);
                g1[4] = 128u;
                g1[5] = (unsigned)Kd;
                g1[6] = 0u;
                g1[7] = 0u;
                asm volatile("tensor_load_to_lds %0, %1" :: "s"(g0), "s"(g1)
                             : "memory");
            }
        } else {
            #pragma unroll
            for (int it = 0; it < 2; ++it) {
                int flat = tid + it * 256;
                int k = flat >> 4, mch = (flat & 15) * 8;
                bf16x8 v = *(const bf16x8*)(A + (size_t)(k0 + k) * M + mBase + mch);
                #pragma unroll
                for (int e = 0; e < 8; ++e) As[buf][mch + e][k] = v[e];
            }
        }
        #pragma unroll
        for (int it = 0; it < 2; ++it) {
            int flat = tid + it * 256;
            int r = flat >> 4, ch = (flat & 15) * 8;
            bf16x8 v = *(const bf16x8*)(Bm + (size_t)(k0 + r) * N + nBase + ch);
            int sk = swzk(r);
            #pragma unroll
            for (int e = 0; e < 8; ++e) Bsw[buf][ch + e][sk] = v[e];
        }
    };

    v8f acc[4][2];
    #pragma unroll
    for (int i = 0; i < 4; ++i)
        #pragma unroll
        for (int j = 0; j < 2; ++j) acc[i][j] = (v8f)0.0f;

    stage(kStart, 0);
    for (int t = 0; t < nT; ++t) {
        int buf = t & 1;
        if (!ATRANS && wave == 0) __builtin_amdgcn_s_wait_tensorcnt(0);
        __syncthreads();                    // tile t ready; prev reads of buf^1 done
        if (t + 1 < nT) {
            stage(kStart + (t + 1) * 32, buf ^ 1);
            if (t + 2 < nT)                 // prefetch B tile t+2 -> global_prefetch_b8
                __builtin_prefetch(
                    Bm + (size_t)(kStart + (t + 2) * 32) * N + nBase + (tid & 15) * 8,
                    0, 0);
        }
        v16bf afr[4];
        #pragma unroll
        for (int mt = 0; mt < 4; ++mt)
            afr[mt] = load_afrag(&As[buf][waveM + mt * 16 + mlane][0], khalf);
        v16bf bfr[2];
        #pragma unroll
        for (int nt = 0; nt < 2; ++nt)
            bfr[nt] = load_bfrag(&Bsw[buf][waveN + nt * 16 + mlane][0], half);
        #pragma unroll
        for (int mt = 0; mt < 4; ++mt)
            #pragma unroll
            for (int nt = 0; nt < 2; ++nt)
                acc[mt][nt] = __builtin_amdgcn_wmma_f32_16x16x32_bf16(
                    false, afr[mt], false, bfr[nt], (short)0, acc[mt][nt],
                    false, false);
    }

    #pragma unroll
    for (int mt = 0; mt < 4; ++mt)
        #pragma unroll
        for (int nt = 0; nt < 2; ++nt) {
            int n = nBase + waveN + nt * 16 + mlane;
            float bv = (EPI == 2) ? bias[n] : 0.0f;
            #pragma unroll
            for (int r = 0; r < 8; ++r) {
                int m = mBase + waveM + mt * 16 + r + half * 8;
                size_t idx = (size_t)m * N + n;
                float v = acc[mt][nt][r] + bv;
                if (EPI == 1) v += C[idx];
                if (EPI == 3) atomicAdd(&C[idx], v);
                else          C[idx] = v;
            }
        }
}

// ---------------------------------------------------------------------------
// AR-term helpers
// ---------------------------------------------------------------------------
__global__ __launch_bounds__(256)
void build_shift(const float* xn, __bf16* ash) {
    int row = blockIdx.x, d = threadIdx.x;
    int l = row & 1023;
    #pragma unroll
    for (int i = 0; i < KU; ++i) {
        float v = (l >= i) ? xn[(size_t)(row - i) * DD + d] : 0.f;
        ash[(size_t)row * (KU * DD) + i * DD + d] = f2bf(v);
    }
}

__global__ __launch_bounds__(256)
void pack_mu(const float* mu, __bf16* wu) {
    int r = blockIdx.x, o = threadIdx.x;
    int iu = r >> 8, din = r & 255;
    wu[(size_t)r * DD + o] = f2bf(mu[(size_t)o * (DD * KU) + din * KU + iu]);
}

__global__ __launch_bounds__(256)
void pack_my(const float* my, __bf16* wy) {
    int r = blockIdx.x, o = threadIdx.x;
    wy[(size_t)r * DD + o] = f2bf(my[(size_t)o * (KYY * DD) + r]);
}

// ---------------------------------------------------------------------------
// Sequential recurrence  y_t = sum_j M_y[:,j,:] y_{t-1-j} + delta_t
// ---------------------------------------------------------------------------
#define CPAD 520
__global__ __launch_bounds__(512)
void scan_y(const __bf16* __restrict__ Wy, const float* __restrict__ delta,
            float* __restrict__ y) {
    __shared__ __bf16 carry[16][CPAD];    // 1040B row stride: conflict-free b128
    __shared__ float  outs[8][DD];
    int tid  = threadIdx.x;
    int lane = tid & 31;
    int wave = tid >> 5;                  // 0..15 -> n-tile
    int n0   = wave * 16;
    int mlane = lane & 15;
    int half  = lane >> 4;
    int khalf = half * 8;

    for (int i = tid; i < 16 * CPAD / 2; i += 512)
        ((unsigned*)&carry[0][0])[i] = 0u;
    __syncthreads();

    // preload constant B fragments (M_y^T): 16 k-tiles of 16x16x32
    v16bf bfr[16];
    #pragma unroll
    for (int kt = 0; kt < 16; ++kt) {
        int kb = kt * 32;
        v16bf b;
        #pragma unroll
        for (int i = 0; i < 8; ++i) {
            int kk = kb + ((i < 4) ? 0 : 16) + khalf + (i & 3) * 2;
            b[2 * i]     = Wy[(size_t)kk * DD + n0 + mlane];
            b[2 * i + 1] = Wy[(size_t)(kk + 1) * DD + n0 + mlane];
        }
        bfr[kt] = b;
    }

    for (int l = 0; l < LL; ++l) {
        v8f acc = (v8f)0.0f;
        #pragma unroll
        for (int kt = 0; kt < 16; ++kt) {
            v16bf a = load_afrag(&carry[mlane][kt * 32], khalf);
            acc = __builtin_amdgcn_wmma_f32_16x16x32_bf16(
                false, a, false, bfr[kt], (short)0, acc, false, false);
        }
        int n = n0 + mlane;
        #pragma unroll
        for (int r = 0; r < 8; ++r) {
            int m = r + half * 8;
            if (m < BB) {
                size_t idx = (size_t)(m * LL + l) * DD + n;
                float v = acc[r] + delta[idx];
                y[idx] = v;
                outs[m][n] = v;
            }
        }
        __syncthreads();
        // shift: ky=1 <- ky=0 ; ky=0 <- new output (32-bit pair moves)
        unsigned t32[2];
        float2   o2[2];
        #pragma unroll
        for (int i = 0; i < 2; ++i) {
            int p = tid + i * 512;        // 1024 pairs of consecutive d
            int bb = p >> 7, dd = (p & 127) * 2;
            t32[i] = *(const unsigned*)&carry[bb][dd];
            o2[i]  = *(const float2*)&outs[bb][dd];
        }
        __syncthreads();
        #pragma unroll
        for (int i = 0; i < 2; ++i) {
            int p = tid + i * 512;
            int bb = p >> 7, dd = (p & 127) * 2;
            *(unsigned*)&carry[bb][DD + dd] = t32[i];
            *(unsigned*)&carry[bb][dd]      = pack2bf(o2[i].x, o2[i].y);
        }
        __syncthreads();
    }
}

// ---------------------------------------------------------------------------
// Pointwise epilogues
// ---------------------------------------------------------------------------
__global__ __launch_bounds__(256)
void gelu_kernel(const float* y, __bf16* act) {
    size_t idx = (size_t)blockIdx.x * DD + threadIdx.x;
    float v = y[idx];
    act[idx] = f2bf(0.5f * v * (1.0f + erff(v * 0.70710678118654752f)));
}

__global__ __launch_bounds__(256)
void glu_residual(const float* glu, const float* z, float* xout) {
    int row = blockIdx.x, d = threadIdx.x;
    float a = glu[(size_t)row * (2 * DD) + d];
    float g = glu[(size_t)row * (2 * DD) + DD + d];
    xout[(size_t)row * DD + d] =
        a * (1.0f / (1.0f + __expf(-g))) + z[(size_t)row * DD + d];
}

__global__ __launch_bounds__(256)
void mean_proj(const float* x, const float* pw, const float* pb, float* out) {
    __shared__ float sm[DD];
    int b = blockIdx.x, t = threadIdx.x;
    float s = 0.f;
    for (int l = 0; l < LL; ++l) s += x[(size_t)(b * LL + l) * DD + t];
    sm[t] = s * (1.0f / (float)LL);
    __syncthreads();
    if (t < DT) {
        float o = pb[t];
        for (int d = 0; d < DD; ++d) o += sm[d] * pw[d * DT + t];
        out[b * DT + t] = o;
    }
}

// ---------------------------------------------------------------------------
// Host launcher
// ---------------------------------------------------------------------------
extern "C" void kernel_launch(void* const* d_in, const int* in_sizes, int n_in,
                              void* d_out, int out_size, void* d_ws, size_t ws_size,
                              hipStream_t stream) {
    const float* inputs   = (const float*)d_in[0];
    const float* emb_w    = (const float*)d_in[1];
    const float* emb_b    = (const float*)d_in[2];
    const float* bn_gamma = (const float*)d_in[3];
    const float* bn_beta  = (const float*)d_in[4];
    const float* m_y      = (const float*)d_in[5];
    const float* m_u      = (const float*)d_in[6];
    const float* m_phi    = (const float*)d_in[7];
    const float* lin_w    = (const float*)d_in[8];
    const float* lin_b    = (const float*)d_in[9];
    const float* proj_w   = (const float*)d_in[10];
    const float* proj_b   = (const float*)d_in[11];
    const float* eig_vals = (const float*)d_in[12];
    const float* eig_vecs = (const float*)d_in[13];
    float* out = (float*)d_out;

    char* ws = (char*)d_ws;
    size_t off = 0;
    auto alloc = [&](size_t bytes) -> void* {
        void* p = ws + off;
        off = (off + bytes + 255) & ~(size_t)255;
        return p;
    };

    float*   xA     = (float*)alloc((size_t)BL * DD * 4);
    float*   xB     = (float*)alloc((size_t)BL * DD * 4);
    float*   xn     = (float*)alloc((size_t)BL * DD * 4);
    float2*  U      = (float2*)alloc((size_t)BB * DD * NFFT * 8);
    float2*  Vf     = (float2*)alloc((size_t)KK * NFFT * 8);
    __bf16*  xtT    = (__bf16*)alloc((size_t)BL * KD * 2);   // [K*D][BL]
    float*   delta  = (float*)alloc((size_t)BL * DD * 4);
    __bf16*  ash    = (__bf16*)alloc((size_t)BL * KU * DD * 2);
    float*   glu    = (float*)alloc((size_t)BL * 2 * DD * 4);
    float*   ybuf   = (float*)alloc((size_t)BL * DD * 4);
    __bf16*  actb   = (__bf16*)alloc((size_t)BL * DD * 2);
    __bf16*  mphi_b = (__bf16*)alloc((size_t)KD * DD * 2);
    __bf16*  mu_b   = (__bf16*)alloc((size_t)KU * DD * DD * 2);
    __bf16*  my_b   = (__bf16*)alloc((size_t)KYY * DD * DD * 2);
    __bf16*  linw_b = (__bf16*)alloc((size_t)DD * 2 * DD * 2);
    float*   stats  = (float*)alloc(2 * DD * 4);
    (void)ws_size; (void)in_sizes; (void)n_in; (void)out_size;

    embed_kernel<<<BL, DD, 0, stream>>>(inputs, emb_w, emb_b, xA);
    fft_filters<<<KK, 256, 0, stream>>>(eig_vecs, eig_vals, Vf);

    float* xin  = xA;
    float* xout = xB;
    for (int i = 0; i < NL; ++i) {
        zero_f32<<<2, 256, 0, stream>>>(stats, 2 * DD);
        bn_stats<<<BL / 256, 256, 0, stream>>>(xin, stats);
        bn_apply<<<BL, DD, 0, stream>>>(xin, stats, bn_gamma + i * DD,
                                        bn_beta + i * DD, xn);
        fft_forward<<<BB * DD, 256, 0, stream>>>(xn, U);
        conv_ifft<<<BB * KK * DD, 256, 0, stream>>>(Vf, U, xtT);

        f32_to_bf16<<<(KD * DD + 255) / 256, 256, 0, stream>>>(
            mphi_b, m_phi + (size_t)i * KD * DD, KD * DD);
        pack_mu<<<KU * DD, DD, 0, stream>>>(m_u + (size_t)i * DD * DD * KU, mu_b);
        pack_my<<<KYY * DD, DD, 0, stream>>>(m_y + (size_t)i * DD * KYY * DD, my_b);
        f32_to_bf16<<<(DD * 2 * DD + 255) / 256, 256, 0, stream>>>(
            linw_b, lin_w + (size_t)i * DD * 2 * DD, DD * 2 * DD);

        // delta = x~ @ m_phi : split-K x4, atomic epilogue (M=8192,N=256,K=6144)
        zero_f32<<<BL * DD / 256, 256, 0, stream>>>(delta, BL * DD);
        gemm_bf16<true, 3><<<dim3(DD / 128, BL / 128, 4), 256, 0, stream>>>(
            xtT, mphi_b, delta, nullptr, BL, DD, KD, KD / 4);
        // delta += shifted(x) @ m_u  (TDM A-tiles; M=8192,N=256,K=768)
        build_shift<<<BL, DD, 0, stream>>>(xn, ash);
        gemm_bf16<false, 1><<<dim3(DD / 128, BL / 128, 1), 256, 0, stream>>>(
            ash, mu_b, delta, nullptr, BL, DD, KU * DD, KU * DD);

        // sequential recurrence (WMMA in the scan loop)
        scan_y<<<1, 512, 0, stream>>>(my_b, delta, ybuf);

        // GELU -> GLU linear (TDM A-tiles) -> GLU + residual
        gelu_kernel<<<BL, DD, 0, stream>>>(ybuf, actb);
        gemm_bf16<false, 2><<<dim3(2 * DD / 128, BL / 128, 1), 256, 0, stream>>>(
            actb, linw_b, glu, lin_b + (size_t)i * 2 * DD, BL, 2 * DD, DD, DD);
        glu_residual<<<BL, DD, 0, stream>>>(glu, xin, xout);

        float* t = xin; xin = xout; xout = t;
    }

    mean_proj<<<BB, DD, 0, stream>>>(xin, proj_w, proj_b, out);
}
